// MaskedMHA_29635274342430
// MI455X (gfx1250) — compile-verified, matrix operand order checked by reading, not run on hardware
//
#include <hip/hip_runtime.h>
#include <stdint.h>

// ---------------------------------------------------------------------------
// MaskedMHA (GQA) for MI455X / gfx1250, wave32, WMMA bf16 16x16x32 f32-acc.
// Shapes: B=4, Q=1024, P=1024, S=2048, H=32, KV=8, D=128, HID=4096, NREP=4.
// d_out layout: out[B,H,Q,D] | k[B,S,KV,D] | v[B,S,KV,D]  (fp32)
// The out region doubles as q scratch (same size); attention reads q first,
// writes out last, per-workgroup read-before-write -> safe and deterministic.
// ---------------------------------------------------------------------------

typedef __attribute__((ext_vector_type(16))) __bf16 v16bf;
typedef __attribute__((ext_vector_type(8)))  __bf16 v8bf;
typedef __attribute__((ext_vector_type(8)))  float  v8f;
typedef __attribute__((ext_vector_type(4)))  int    v4i;

#define GLOBAL_AS __attribute__((address_space(1)))
#define LDS_AS    __attribute__((address_space(3)))

__device__ __forceinline__ v8f vzero8() {
  v8f z = {0.f,0.f,0.f,0.f,0.f,0.f,0.f,0.f};
  return z;
}

#define WMMA_BF16(A, Bm, C) \
  __builtin_amdgcn_wmma_f32_16x16x32_bf16(false, (A), false, (Bm), (short)0, (C), false, false)

// --- async global->LDS copy (16B per lane), ASYNCcnt tracked -----------------
__device__ __forceinline__ void async_cp16(const void* g, void* l) {
#if defined(__has_builtin) && __has_builtin(__builtin_amdgcn_global_load_async_to_lds_b128)
  GLOBAL_AS v4i* gp = (GLOBAL_AS v4i*)(const_cast<void*>(g));
  LDS_AS    v4i* lp = (LDS_AS v4i*)(l);
  __builtin_amdgcn_global_load_async_to_lds_b128(gp, lp, 0, 0);
#else
  *(float4*)l = *(const float4*)g;
#endif
}

__device__ __forceinline__ void wait_async0() {
#if defined(__has_builtin) && __has_builtin(__builtin_amdgcn_s_wait_asynccnt)
  __builtin_amdgcn_s_wait_asynccnt(0);
#else
  asm volatile("s_wait_asynccnt 0" ::: "memory");
#endif
}

// --- fragment loaders --------------------------------------------------------
// 16-bit A/B WMMA layout: lanes 0-15 hold K in {k0..k0+7} u {k0+16..k0+23}
// with k0=0; lanes 16-31 with k0=8. Caller passes p = rowbase + kchunk + k0.
__device__ __forceinline__ v16bf frag_ld_f32(const float* p) {
  v16bf r;
#pragma unroll
  for (int j = 0; j < 8; ++j) {
    r[j]     = (__bf16)p[j];
    r[j + 8] = (__bf16)p[j + 16];
  }
  return r;
}

__device__ __forceinline__ v16bf frag_ld_bf16(const __bf16* p) {
  v8bf lo = *(const v8bf*)(p);
  v8bf hi = *(const v8bf*)(p + 16);
  v16bf r;
#pragma unroll
  for (int j = 0; j < 8; ++j) { r[j] = lo[j]; r[j + 8] = hi[j]; }
  return r;
}

// ---------------------------------------------------------------------------
// Kernel 0: copy KV caches into the first P slots of k/v outputs.
// ---------------------------------------------------------------------------
__global__ __launch_bounds__(256) void copy_cache_kernel(
    const float4* __restrict__ kc, const float4* __restrict__ vc,
    float4* __restrict__ ko, float4* __restrict__ vo) {
  size_t i = (size_t)blockIdx.x * 256 + threadIdx.x;  // 1,048,576 total
  int    b = (int)(i >> 18);                          // P*KV*D/4 = 262144 per b
  size_t r = i & 262143;
  size_t dst = (size_t)b * 524288 + r;                // S*KV*D/4 = 524288 per b
  ko[dst] = kc[i];
  vo[dst] = vc[i];
}

// ---------------------------------------------------------------------------
// Kernel 1: fused QKV GEMM. C[4096,6144] = X[4096,4096] * W[6144,4096]^T + b
// 128x128 tiles, Kstep=32, bf16 WMMA, async double-buffered LDS staging.
// ---------------------------------------------------------------------------
__global__ __launch_bounds__(256) void qkv_gemm_kernel(
    const float* __restrict__ X, const float* __restrict__ W,
    const float* __restrict__ bias,
    float* __restrict__ qout, float* __restrict__ kout, float* __restrict__ vout) {
  constexpr int AS = 36;  // f32 row stride (32 data + 4 pad), 16B aligned
  __shared__ float aT[2][128 * AS];
  __shared__ float bT[2][128 * AS];

  const int mt = blockIdx.y, nt = blockIdx.x;
  const int t = threadIdx.x, lane = t & 31, wid = t >> 5;
  const int wm = wid & 3;   // 4 wave-rows  x 32 M
  const int wn = wid >> 2;  // 2 wave-cols  x 64 N
  const int r15 = lane & 15, k0l = (lane >> 4) << 3, half8 = k0l;

  auto stage = [&](int buf, int k0) {
#pragma unroll
    for (int it = 0; it < 4; ++it) {
      int idx = it * 256 + t;               // 0..1023
      int row = idx >> 3, c4 = (idx & 7) * 4;
      async_cp16(X + (size_t)(mt * 128 + row) * 4096 + k0 + c4,
                 &aT[buf][row * AS + c4]);
      async_cp16(W + (size_t)(nt * 128 + row) * 4096 + k0 + c4,
                 &bT[buf][row * AS + c4]);
    }
  };

  v8f acc[2][4];
#pragma unroll
  for (int i = 0; i < 2; ++i)
#pragma unroll
    for (int j = 0; j < 4; ++j) acc[i][j] = vzero8();

  stage(0, 0);
  wait_async0();
  __syncthreads();

  for (int kk = 0; kk < 128; ++kk) {  // 4096 / 32
    int cur = kk & 1;
    if (kk + 1 < 128) stage(cur ^ 1, (kk + 1) * 32);

    v16bf a[2], bm[4];
#pragma unroll
    for (int i = 0; i < 2; ++i)
      a[i] = frag_ld_f32(&aT[cur][(wm * 32 + i * 16 + r15) * AS + k0l]);
#pragma unroll
    for (int j = 0; j < 4; ++j)
      bm[j] = frag_ld_f32(&bT[cur][(wn * 64 + j * 16 + r15) * AS + k0l]);
#pragma unroll
    for (int i = 0; i < 2; ++i)
#pragma unroll
      for (int j = 0; j < 4; ++j) acc[i][j] = WMMA_BF16(a[i], bm[j], acc[i][j]);

    wait_async0();
    __syncthreads();
  }

  // Epilogue: scatter to q / k_new / v_new regions.
#pragma unroll
  for (int i = 0; i < 2; ++i)
#pragma unroll
    for (int j = 0; j < 4; ++j) {
      int ncol = nt * 128 + wn * 64 + j * 16 + r15;
      float bv = bias[ncol];
#pragma unroll
      for (int e = 0; e < 8; ++e) {
        int r = mt * 128 + wm * 32 + i * 16 + e + half8;
        int bI = r >> 10, q = r & 1023;
        float val = acc[i][j][e] + bv;
        int d = ncol & 127;
        if (ncol < 4096) {
          int h = ncol >> 7;
          qout[(size_t)(((bI * 32 + h) << 10) + q) * 128 + d] = val;
        } else if (ncol < 5120) {
          int h = (ncol - 4096) >> 7;
          kout[(size_t)((bI * 2048 + 1024 + q) * 8 + h) * 128 + d] = val;
        } else {
          int h = (ncol - 5120) >> 7;
          vout[(size_t)((bI * 2048 + 1024 + q) * 8 + h) * 128 + d] = val;
        }
      }
    }
}

// ---------------------------------------------------------------------------
// Kernel 2: GQA flash attention. WG = (b, kvh, 32 q rows); 8 waves =
// 4 reps x 2 q-subtiles sharing K/V LDS tiles. S chunk = 32, online softmax.
// ---------------------------------------------------------------------------
__global__ __launch_bounds__(256) void attn_kernel(
    const float* Qin, const float* __restrict__ Kall,
    const float* __restrict__ Vall, const float* __restrict__ mask,
    float* Out) {
  constexpr int KS = 132;  // f32 stride for K chunk rows [s][d]
  constexpr int VS = 40;   // bf16 stride for V^T rows [d][s] and P rows [q][s]
  __shared__ float  kt[2][32 * KS];    // 33 KB
  __shared__ __bf16 vt[2][128 * VS];   // 20 KB, transposed V (d-major)
  __shared__ __bf16 pt[8][16 * VS];    // 10 KB, per-wave P tiles

  const int b = blockIdx.z, kvh = blockIdx.y, q0 = blockIdx.x * 32;
  const int t = threadIdx.x, lane = t & 31, wid = t >> 5;
  const int rep = wid & 3, qs = wid >> 2;
  const int h = kvh * 4 + rep;
  const int qbase = q0 + qs * 16;
  const int r15 = lane & 15, k0l = (lane >> 4) << 3, half8 = k0l;

  // Load this wave's 16x128 Q tile once (fp32 -> bf16 A-fragments).
  v16bf qf[4];
  {
    const float* qrow =
        Qin + (size_t)((b * 32 + h) * 1024 + qbase + r15) * 128;
#pragma unroll
    for (int kk = 0; kk < 4; ++kk) qf[kk] = frag_ld_f32(qrow + kk * 32 + k0l);
  }

  auto stageK = [&](int buf, int sbase) {
#pragma unroll
    for (int it = 0; it < 4; ++it) {
      int idx = it * 256 + t;  // 0..1023
      int s = idx >> 5, c4 = (idx & 31) * 4;
      async_cp16(Kall + (size_t)((b * 2048 + sbase + s) * 8 + kvh) * 128 + c4,
                 &kt[buf][s * KS + c4]);
    }
  };
  auto stageV = [&](int buf, int sbase) {  // transpose + bf16 convert
#pragma unroll
    for (int it = 0; it < 4; ++it) {
      int idx = it * 256 + t;
      int s = idx >> 5, c4 = (idx & 31) * 4;
      float4 v4 = *(const float4*)(
          Vall + (size_t)((b * 2048 + sbase + s) * 8 + kvh) * 128 + c4);
      vt[buf][(c4 + 0) * VS + s] = (__bf16)v4.x;
      vt[buf][(c4 + 1) * VS + s] = (__bf16)v4.y;
      vt[buf][(c4 + 2) * VS + s] = (__bf16)v4.z;
      vt[buf][(c4 + 3) * VS + s] = (__bf16)v4.w;
    }
  };

  float mrow[8], lrow[8];
  v8f accd[8];
#pragma unroll
  for (int e = 0; e < 8; ++e) { mrow[e] = -3.0e38f; lrow[e] = 0.f; }
#pragma unroll
  for (int nd = 0; nd < 8; ++nd) accd[nd] = vzero8();

  stageK(0, 0);
  stageV(0, 0);
  wait_async0();
  __syncthreads();

  const float sscale = 0.08838834764831845f;  // 1/sqrt(128)

  for (int sc = 0; sc < 64; ++sc) {  // S=2048, chunk 32
    int cur = sc & 1, sb = sc * 32;
    if (sc + 1 < 64) {
      stageK(cur ^ 1, (sc + 1) * 32);
      stageV(cur ^ 1, (sc + 1) * 32);
    }

    // scores = Q * K^T : two 16x16 tiles over this 32-wide s chunk
    v8f s0 = vzero8(), s1 = vzero8();
#pragma unroll
    for (int kk = 0; kk < 4; ++kk) {
      v16bf b0 = frag_ld_f32(&kt[cur][(r15)      * KS + kk * 32 + k0l]);
      v16bf b1 = frag_ld_f32(&kt[cur][(16 + r15) * KS + kk * 32 + k0l]);
      s0 = WMMA_BF16(qf[kk], b0, s0);
      s1 = WMMA_BF16(qf[kk], b1, s1);
    }

    // online softmax; C layout: row = e + 8*(lane>=16), col = ns*16 + (lane&15)
#pragma unroll
    for (int e = 0; e < 8; ++e) {
      int qg = qbase + e + half8;
      const float* mp = mask + (size_t)(b * 1024 + qg) * 2048 + sb;
      float x0 = s0[e] * sscale + mp[r15];
      float x1 = s1[e] * sscale + mp[16 + r15];
      float mx = fmaxf(x0, x1);
#pragma unroll
      for (int off = 1; off < 16; off <<= 1)
        mx = fmaxf(mx, __shfl_xor(mx, off, 32));
      float mn = fmaxf(mrow[e], mx);
      float p0 = __expf(x0 - mn), p1 = __expf(x1 - mn);
      float ssum = p0 + p1;
#pragma unroll
      for (int off = 1; off < 16; off <<= 1) ssum += __shfl_xor(ssum, off, 32);
      float corr = __expf(mrow[e] - mn);
      lrow[e] = lrow[e] * corr + ssum;
      mrow[e] = mn;
#pragma unroll
      for (int nd = 0; nd < 8; ++nd) accd[nd][e] *= corr;
      int prow = e + half8;
      pt[wid][prow * VS + r15]      = (__bf16)p0;
      pt[wid][prow * VS + 16 + r15] = (__bf16)p1;
    }

    // O += P[16x32] * V[32x128]
    v16bf pf = frag_ld_bf16(&pt[wid][r15 * VS + k0l]);
#pragma unroll
    for (int nd = 0; nd < 8; ++nd) {
      v16bf vf = frag_ld_bf16(&vt[cur][(nd * 16 + r15) * VS + k0l]);
      accd[nd] = WMMA_BF16(pf, vf, accd[nd]);
    }

    wait_async0();
    __syncthreads();
  }

  // normalize + write out (overwrites the q scratch rows this WG owns)
#pragma unroll
  for (int e = 0; e < 8; ++e) {
    float inv = 1.0f / lrow[e];
    int qg = qbase + e + half8;
    float* orow = Out + (size_t)((b * 32 + h) * 1024 + qg) * 128;
#pragma unroll
    for (int nd = 0; nd < 8; ++nd) orow[nd * 16 + r15] = accd[nd][e] * inv;
  }
}

// ---------------------------------------------------------------------------
extern "C" void kernel_launch(void* const* d_in, const int* in_sizes, int n_in,
                              void* d_out, int out_size, void* d_ws,
                              size_t ws_size, hipStream_t stream) {
  (void)in_sizes; (void)n_in; (void)out_size; (void)d_ws; (void)ws_size;
  const float* input = (const float*)d_in[0];
  const float* kc    = (const float*)d_in[1];
  const float* vc    = (const float*)d_in[2];
  // d_in[3] = max_position (unused), d_in[5] = beam_idx (unused by reference)
  const float* mask  = (const float*)d_in[4];
  const float* w     = (const float*)d_in[6];
  const float* bias  = (const float*)d_in[7];

  float* out  = (float*)d_out;                       // [4,32,1024,128]
  float* kout = out + (size_t)4 * 32 * 1024 * 128;   // [4,2048,8,128]
  float* vout = kout + (size_t)4 * 2048 * 8 * 128;   // [4,2048,8,128]

  copy_cache_kernel<<<dim3(4096), dim3(256), 0, stream>>>(
      (const float4*)kc, (const float4*)vc, (float4*)kout, (float4*)vout);
  qkv_gemm_kernel<<<dim3(48, 32), dim3(256), 0, stream>>>(
      input, w, bias, out, kout, vout);
  attn_kernel<<<dim3(32, 8, 4), dim3(256), 0, stream>>>(
      out, kout, vout, mask, out);
}
